// LlamaMoEDecoderLayer_15307263443375
// MI455X (gfx1250) — matrix operand order repaired and checked by reading, not compiled, not run on hardware
//
#include <hip/hip_runtime.h>
#include <hip/hip_bf16.h>

typedef __bf16 bf16;
typedef __attribute__((ext_vector_type(16))) __bf16 v16bf;
typedef __attribute__((ext_vector_type(8)))  __bf16 v8bf;
typedef __attribute__((ext_vector_type(8)))  float  v8f;

#define B_  2
#define S_  1024
#define D_  2048
#define H_  16
#define HD_ 128
#define E_  8
#define DE_ 704
#define N_  (B_ * S_)
#define EPS_ 1e-6f

// ---------------- WMMA fragment helpers (CDNA5 wave32 layouts) ----------------
// A 16x32 bf16 from row-major [M,K] tile: lane=M row; half-waves split K
// {0-7,16-23} / {8-15,24-31}.
__device__ __forceinline__ v16bf load_fragA(const bf16* p, int ld) {
  int lane = threadIdx.x & 31;
  const bf16* q = p + (size_t)(lane & 15) * ld + ((lane >> 4) << 3);
  union { v16bf v; v8bf h[2]; } u;
  u.h[0] = *(const v8bf*)(q);
  u.h[1] = *(const v8bf*)(q + 16);
  return u.v;
}
// B 32x16 bf16 from row-major [N,K] tile (i.e. W[out,in]): lane=N col;
// half-waves split K {0-7,16-23} / {8-15,24-31}.
__device__ __forceinline__ v16bf load_fragB(const bf16* p, int ld) {
  int lane = threadIdx.x & 31;
  const bf16* q = p + (size_t)(lane & 15) * ld + ((lane >> 4) << 3);
  union { v16bf v; v8bf h[2]; } u;
  u.h[0] = *(const v8bf*)(q);
  u.h[1] = *(const v8bf*)(q + 16);
  return u.v;
}
__device__ __forceinline__ v8f wmma_bf(v16bf a, v16bf b, v8f c) {
  return __builtin_amdgcn_wmma_f32_16x16x32_bf16(false, a, false, b, (short)0, c,
                                                 false, false);
}
#define VZERO8 {0.f,0.f,0.f,0.f,0.f,0.f,0.f,0.f}
#define VZERO2x2 {{VZERO8, VZERO8}, {VZERO8, VZERO8}}
#define VZERO2x4 {{VZERO8, VZERO8, VZERO8, VZERO8}, {VZERO8, VZERO8, VZERO8, VZERO8}}

// ---------------- fp32 -> bf16 conversion ----------------
__global__ void k_cvt(const float* __restrict__ src, bf16* __restrict__ dst, int n) {
  int i = blockIdx.x * 256 + threadIdx.x;
  if (i < n) dst[i] = (bf16)src[i];
}

// ---------------- RMSNorm -> bf16 ----------------
__global__ void k_rmsnorm(const float* __restrict__ x, const float* __restrict__ w,
                          bf16* __restrict__ out) {
  __shared__ float red[256];
  int row = blockIdx.x;
  const float* xr = x + (size_t)row * D_;
  float s = 0.f;
  for (int d = threadIdx.x; d < D_; d += 256) { float v = xr[d]; s += v * v; }
  red[threadIdx.x] = s;
  __syncthreads();
  for (int o = 128; o > 0; o >>= 1) {
    if (threadIdx.x < (unsigned)o) red[threadIdx.x] += red[threadIdx.x + o];
    __syncthreads();
  }
  float inv = rsqrtf(red[0] / (float)D_ + EPS_);
  for (int d = threadIdx.x; d < D_; d += 256)
    out[(size_t)row * D_ + d] = (bf16)(xr[d] * inv * w[d]);
}

// ------- Wide GEMM: C[M,N] = A[M,K](bf16) * W[N,K]^T(bf16), N % 128 == 0 -------
// block=128 (4 waves as 2x2), block tile 64x128, wave tile 32x64 (2x4 frags):
// 12 b128 loads per 8 WMMAs.
__global__ void k_gemm_w(const bf16* __restrict__ A, int lda,
                         const bf16* __restrict__ W, int ldw,
                         float* __restrict__ Cf, bf16* __restrict__ Cb, int ldc,
                         const float* __restrict__ resid, int K) {
  int wave = threadIdx.x >> 5, lane = threadIdx.x & 31;
  int row0 = blockIdx.y * 64 + (wave >> 1) * 32;
  int col0 = blockIdx.x * 128 + (wave & 1) * 64;
  v8f acc[2][4] = VZERO2x4;
  const bf16* Ar0 = A + (size_t)row0 * lda;
  const bf16* Ar1 = A + (size_t)(row0 + 16) * lda;
  const bf16* Wc[4];
#pragma unroll
  for (int ni = 0; ni < 4; ni++) Wc[ni] = W + (size_t)(col0 + ni * 16) * ldw;
  for (int k = 0; k < K; k += 32) {
    if (k + 32 < K) {  // prefetch next K-tile of the streaming weight operand
      __builtin_prefetch(Wc[0] + (size_t)(lane & 15) * ldw + k + 32, 0, 1);
      __builtin_prefetch(Wc[2] + (size_t)(lane & 15) * ldw + k + 32, 0, 1);
    }
    v16bf a0 = load_fragA(Ar0 + k, lda);
    v16bf a1 = load_fragA(Ar1 + k, lda);
#pragma unroll
    for (int ni = 0; ni < 4; ni++) {
      v16bf b = load_fragB(Wc[ni] + k, ldw);
      acc[0][ni] = wmma_bf(a0, b, acc[0][ni]);
      acc[1][ni] = wmma_bf(a1, b, acc[1][ni]);
    }
  }
  int n = lane & 15, mb = (lane >> 4) * 8;
#pragma unroll
  for (int mi = 0; mi < 2; mi++)
#pragma unroll
    for (int ni = 0; ni < 4; ni++)
#pragma unroll
      for (int i = 0; i < 8; i++) {
        size_t r = row0 + mi * 16 + mb + i, c = col0 + ni * 16 + n;
        float v = acc[mi][ni][i];
        if (resid) v += resid[r * ldc + c];
        if (Cb) Cb[r * ldc + c] = (bf16)v;
        else    Cf[r * ldc + c] = v;
      }
}

// ---------------- RoPE + bf16 pack; V stored transposed ----------------
__global__ void k_rope(const float* __restrict__ qf, const float* __restrict__ kf,
                       const float* __restrict__ vf, const int* __restrict__ pos_ids,
                       bf16* __restrict__ qb, bf16* __restrict__ kb,
                       bf16* __restrict__ vtb) {
  int tok = blockIdx.x;           // b*S + s
  int h   = blockIdx.y;
  int d   = threadIdx.x;          // 0..127
  int b = tok / S_, s = tok - b * S_;
  float pos = (float)pos_ids[tok];
  int i = d & 63;
  float invf = expf(-9.210340371976184f * (float)i * (1.0f / 64.0f)); // theta^(-2i/HD)
  float ang = pos * invf;
  float c = cosf(ang), sn = sinf(ang);
  size_t src = (size_t)tok * D_ + h * HD_ + d;
  size_t srcr = (size_t)tok * D_ + h * HD_ + ((d < 64) ? d + 64 : d - 64);
  float qv = qf[src], kv = kf[src], vv = vf[src];
  float qr = (d < 64) ? -qf[srcr] : qf[srcr];
  float kr = (d < 64) ? -kf[srcr] : kf[srcr];
  float qo = (qv * c + qr * sn) * 0.08838834764831845f;  // fold 1/sqrt(HD)
  float ko = kv * c + kr * sn;
  size_t bh = (size_t)b * H_ + h;
  qb[(bh * S_ + s) * HD_ + d] = (bf16)qo;
  kb[(bh * S_ + s) * HD_ + d] = (bf16)ko;
  vtb[(bh * HD_ + d) * S_ + s] = (bf16)vv;
}

// ---------------- Flash attention (causal, online softmax), all-WMMA ----------------
// grid (S/64, B*H), block 128: wave w owns 16 query rows; BN=32 keys/iter.
__global__ void k_flash(const bf16* __restrict__ Q, const bf16* __restrict__ Km,
                        const bf16* __restrict__ Vt, bf16* __restrict__ Ob) {
  __shared__ bf16 pbuf[4][16 * 32];
  int wave = threadIdx.x >> 5, lane = threadIdx.x & 31;
  int bh = blockIdx.y;
  int b = bh / H_, h = bh - b * H_;
  int q0 = blockIdx.x * 64 + wave * 16;
  const bf16* Qbh = Q + (size_t)bh * S_ * HD_;
  const bf16* Kbh = Km + (size_t)bh * S_ * HD_;
  const bf16* Vbh = Vt + (size_t)bh * HD_ * S_;
  int n = lane & 15, mb = (lane >> 4) * 8;

  v16bf qa[4];
#pragma unroll
  for (int j = 0; j < 4; j++) qa[j] = load_fragA(Qbh + (size_t)q0 * HD_ + 32 * j, HD_);

  v8f of[8] = {VZERO8, VZERO8, VZERO8, VZERO8, VZERO8, VZERO8, VZERO8, VZERO8};
  float m_i[8], l_i[8];
#pragma unroll
  for (int i = 0; i < 8; i++) { m_i[i] = -3.0e38f; l_i[i] = 0.f; }

  for (int kb = 0; kb < q0 + 16; kb += 32) {
    v8f s0 = VZERO8, s1 = VZERO8;
#pragma unroll
    for (int j = 0; j < 4; j++) {
      s0 = wmma_bf(qa[j], load_fragB(Kbh + (size_t)kb * HD_ + 32 * j, HD_), s0);
      s1 = wmma_bf(qa[j], load_fragB(Kbh + (size_t)(kb + 16) * HD_ + 32 * j, HD_), s1);
    }
    if (kb + 32 > q0) {  // diagonal block: causal mask
#pragma unroll
      for (int i = 0; i < 8; i++) {
        int qrow = q0 + mb + i;
        if (kb + n > qrow)      s0[i] = -1e30f;
        if (kb + 16 + n > qrow) s1[i] = -1e30f;
      }
    }
    float p0[8], p1[8], alpha[8];
#pragma unroll
    for (int i = 0; i < 8; i++) {
      float mx = fmaxf(s0[i], s1[i]);
#pragma unroll
      for (int o = 1; o < 16; o <<= 1) mx = fmaxf(mx, __shfl_xor(mx, o));
      float mnew = fmaxf(m_i[i], mx);
      alpha[i] = __expf(m_i[i] - mnew);
      p0[i] = __expf(s0[i] - mnew);
      p1[i] = __expf(s1[i] - mnew);
      float r = p0[i] + p1[i];
#pragma unroll
      for (int o = 1; o < 16; o <<= 1) r += __shfl_xor(r, o);
      l_i[i] = l_i[i] * alpha[i] + r;
      m_i[i] = mnew;
    }
#pragma unroll
    for (int t = 0; t < 8; t++)
#pragma unroll
      for (int i = 0; i < 8; i++) of[t][i] *= alpha[i];
    // stage P (C-layout) -> LDS row-major -> A-layout frag; same-wave LDS is in-order
    bf16* pb = pbuf[wave];
#pragma unroll
    for (int i = 0; i < 8; i++) {
      pb[(mb + i) * 32 + n]      = (bf16)p0[i];
      pb[(mb + i) * 32 + 16 + n] = (bf16)p1[i];
    }
    v16bf pa = load_fragA(pb, 32);
#pragma unroll
    for (int t = 0; t < 8; t++)
      of[t] = wmma_bf(pa, load_fragB(Vbh + (size_t)(t * 16) * S_ + kb, S_), of[t]);
  }
#pragma unroll
  for (int t = 0; t < 8; t++)
#pragma unroll
    for (int i = 0; i < 8; i++) {
      int qrow = q0 + mb + i;
      Ob[((size_t)b * S_ + qrow) * D_ + h * HD_ + t * 16 + n] =
          (bf16)(of[t][i] / l_i[i]);
    }
}

// ---------------- Router: logits, top-2, softmax weights ----------------
__global__ void k_router(const bf16* __restrict__ xn, const float* __restrict__ gw,
                         float* __restrict__ wexp) {
  __shared__ float red[E_ * 256];
  int row = blockIdx.x, tid = threadIdx.x;
  float acc[E_];
#pragma unroll
  for (int e = 0; e < E_; e++) acc[e] = 0.f;
  const bf16* xr = xn + (size_t)row * D_;
  for (int d = tid; d < D_; d += 256) {
    float xv = (float)xr[d];
#pragma unroll
    for (int e = 0; e < E_; e++) acc[e] += xv * gw[e * D_ + d];
  }
#pragma unroll
  for (int e = 0; e < E_; e++) red[e * 256 + tid] = acc[e];
  __syncthreads();
  for (int o = 128; o > 0; o >>= 1) {
    if (tid < (unsigned)o)
#pragma unroll
      for (int e = 0; e < E_; e++) red[e * 256 + tid] += red[e * 256 + tid + o];
    __syncthreads();
  }
  if (tid == 0) {
    float l[E_];
#pragma unroll
    for (int e = 0; e < E_; e++) l[e] = red[e * 256];
    int i1 = 0;
    for (int e = 1; e < E_; e++) if (l[e] > l[i1]) i1 = e;   // ties -> lowest idx
    int i2 = (i1 == 0) ? 1 : 0;
    for (int e = 0; e < E_; e++) if (e != i1 && l[e] > l[i2]) i2 = e;
    float w1 = 1.f / (1.f + expf(l[i2] - l[i1]));
#pragma unroll
    for (int e = 0; e < E_; e++) wexp[(size_t)row * E_ + e] = 0.f;
    wexp[(size_t)row * E_ + i1] = w1;
    wexp[(size_t)row * E_ + i2] = 1.f - w1;
  }
}

// ---------------- MoE gate/up fused GEMM + SiLU*Up epilogue, per-tile skip ----------------
__global__ void k_moe_gu(const bf16* __restrict__ X, const bf16* __restrict__ Wg,
                         const bf16* __restrict__ Wu, const float* __restrict__ wexp,
                         bf16* __restrict__ inter) {
  int e = blockIdx.z;
  int wave = threadIdx.x >> 5, lane = threadIdx.x & 31;
  int row0 = blockIdx.y * 64 + (wave >> 1) * 32;
  int col0 = blockIdx.x * 64 + (wave & 1) * 32;
  int n = lane & 15, mb = (lane >> 4) * 8;
  bf16* ie = inter + (size_t)e * N_ * DE_;
  bool any = false;
#pragma unroll
  for (int mi = 0; mi < 2; mi++)
#pragma unroll
    for (int i = 0; i < 8; i++)
      any |= (wexp[(size_t)(row0 + mi * 16 + mb + i) * E_ + e] != 0.f);
  if (__ballot(any) == 0ull) {  // no token in this 32-row group uses expert e
#pragma unroll
    for (int mi = 0; mi < 2; mi++)
#pragma unroll
      for (int ni = 0; ni < 2; ni++)
#pragma unroll
        for (int i = 0; i < 8; i++)
          ie[(size_t)(row0 + mi * 16 + mb + i) * DE_ + col0 + ni * 16 + n] = (bf16)0.f;
    return;
  }
  const bf16* wg = Wg + (size_t)e * DE_ * D_;
  const bf16* wu = Wu + (size_t)e * DE_ * D_;
  v8f g[2][2] = VZERO2x2;
  v8f u[2][2] = VZERO2x2;
  for (int k = 0; k < D_; k += 32) {
    v16bf a0 = load_fragA(X + (size_t)row0 * D_ + k, D_);
    v16bf a1 = load_fragA(X + (size_t)(row0 + 16) * D_ + k, D_);
    v16bf bg0 = load_fragB(wg + (size_t)col0 * D_ + k, D_);
    v16bf bg1 = load_fragB(wg + (size_t)(col0 + 16) * D_ + k, D_);
    v16bf bu0 = load_fragB(wu + (size_t)col0 * D_ + k, D_);
    v16bf bu1 = load_fragB(wu + (size_t)(col0 + 16) * D_ + k, D_);
    g[0][0] = wmma_bf(a0, bg0, g[0][0]);  g[0][1] = wmma_bf(a0, bg1, g[0][1]);
    g[1][0] = wmma_bf(a1, bg0, g[1][0]);  g[1][1] = wmma_bf(a1, bg1, g[1][1]);
    u[0][0] = wmma_bf(a0, bu0, u[0][0]);  u[0][1] = wmma_bf(a0, bu1, u[0][1]);
    u[1][0] = wmma_bf(a1, bu0, u[1][0]);  u[1][1] = wmma_bf(a1, bu1, u[1][1]);
  }
#pragma unroll
  for (int mi = 0; mi < 2; mi++)
#pragma unroll
    for (int ni = 0; ni < 2; ni++)
#pragma unroll
      for (int i = 0; i < 8; i++) {
        float gv = g[mi][ni][i];
        float val = (gv / (1.f + __expf(-gv))) * u[mi][ni][i];  // silu(g)*u
        ie[(size_t)(row0 + mi * 16 + mb + i) * DE_ + col0 + ni * 16 + n] = (bf16)val;
      }
}

// ---------------- MoE down GEMM: loop experts, per-row weight, +residual ----------------
__global__ void k_moe_down(const bf16* __restrict__ inter, const bf16* __restrict__ Wd,
                           const float* __restrict__ wexp, const float* __restrict__ resid,
                           float* __restrict__ out) {
  int wave = threadIdx.x >> 5, lane = threadIdx.x & 31;
  int row0 = blockIdx.y * 64 + (wave >> 1) * 32;
  int col0 = blockIdx.x * 64 + (wave & 1) * 32;
  int n = lane & 15, mb = (lane >> 4) * 8;
  v8f acc[2][2] = VZERO2x2;
  for (int e = 0; e < E_; e++) {
    float wrow[2][8];
    bool any = false;
#pragma unroll
    for (int mi = 0; mi < 2; mi++)
#pragma unroll
      for (int i = 0; i < 8; i++) {
        wrow[mi][i] = wexp[(size_t)(row0 + mi * 16 + mb + i) * E_ + e];
        any |= (wrow[mi][i] != 0.f);
      }
    if (__ballot(any) == 0ull) continue;
    const bf16* ie = inter + (size_t)e * N_ * DE_;
    const bf16* wd = Wd + (size_t)e * D_ * DE_;
    v8f t[2][2] = VZERO2x2;
    for (int k = 0; k < DE_; k += 32) {
      v16bf a0 = load_fragA(ie + (size_t)row0 * DE_ + k, DE_);
      v16bf a1 = load_fragA(ie + (size_t)(row0 + 16) * DE_ + k, DE_);
      v16bf b0 = load_fragB(wd + (size_t)col0 * DE_ + k, DE_);
      v16bf b1 = load_fragB(wd + (size_t)(col0 + 16) * DE_ + k, DE_);
      t[0][0] = wmma_bf(a0, b0, t[0][0]);  t[0][1] = wmma_bf(a0, b1, t[0][1]);
      t[1][0] = wmma_bf(a1, b0, t[1][0]);  t[1][1] = wmma_bf(a1, b1, t[1][1]);
    }
#pragma unroll
    for (int mi = 0; mi < 2; mi++)
#pragma unroll
      for (int ni = 0; ni < 2; ni++)
#pragma unroll
        for (int i = 0; i < 8; i++)
          acc[mi][ni][i] += wrow[mi][i] * t[mi][ni][i];
  }
#pragma unroll
  for (int mi = 0; mi < 2; mi++)
#pragma unroll
    for (int ni = 0; ni < 2; ni++)
#pragma unroll
      for (int i = 0; i < 8; i++) {
        size_t r = row0 + mi * 16 + mb + i, c = col0 + ni * 16 + n;
        out[r * D_ + c] = resid[r * D_ + c] + acc[mi][ni][i];
      }
}

// ---------------- host launcher ----------------
extern "C" void kernel_launch(void* const* d_in, const int* in_sizes, int n_in,
                              void* d_out, int out_size, void* d_ws, size_t ws_size,
                              hipStream_t stream) {
  const float* hidden = (const float*)d_in[0];
  const float* ln1    = (const float*)d_in[1];
  const float* ln2    = (const float*)d_in[2];
  const float* qw     = (const float*)d_in[3];
  const float* kw     = (const float*)d_in[4];
  const float* vw     = (const float*)d_in[5];
  const float* ow     = (const float*)d_in[6];
  const float* gatew  = (const float*)d_in[7];
  const float* egw    = (const float*)d_in[8];
  const float* euw    = (const float*)d_in[9];
  const float* edw    = (const float*)d_in[10];
  const int*   pos    = (const int*)d_in[11];
  float* out = (float*)d_out;

  char* ws = (char*)d_ws;
  size_t off = 0;
  auto alloc = [&](size_t bytes) -> void* {
    off = (off + 255) & ~(size_t)255;
    void* p = ws + off;
    off += bytes;
    return p;
  };
  const size_t DD = (size_t)D_ * D_, EW = (size_t)E_ * DE_ * D_;
  bf16* wq  = (bf16*)alloc(DD * 2);
  bf16* wk  = (bf16*)alloc(DD * 2);
  bf16* wv  = (bf16*)alloc(DD * 2);
  bf16* wo  = (bf16*)alloc(DD * 2);
  bf16* weg = (bf16*)alloc(EW * 2);
  bf16* weu = (bf16*)alloc(EW * 2);
  bf16* wed = (bf16*)alloc(EW * 2);
  bf16* xn1 = (bf16*)alloc((size_t)N_ * D_ * 2);
  float* qf = (float*)alloc((size_t)N_ * D_ * 4);
  float* kf = (float*)alloc((size_t)N_ * D_ * 4);
  float* vf = (float*)alloc((size_t)N_ * D_ * 4);
  bf16* qb  = (bf16*)alloc((size_t)N_ * D_ * 2);
  bf16* kb  = (bf16*)alloc((size_t)N_ * D_ * 2);
  bf16* vtb = (bf16*)alloc((size_t)N_ * D_ * 2);
  bf16* atb = (bf16*)alloc((size_t)N_ * D_ * 2);
  bf16* xn2 = (bf16*)alloc((size_t)N_ * D_ * 2);
  float* wgt = (float*)alloc((size_t)N_ * E_ * 4);
  // reuse: qf region (16.8MB) -> h (fp32, 16.8MB); kf+vf region (33.5MB) -> inter (23.1MB)
  float* hf  = qf;
  bf16* inter = (bf16*)kf;

  // 1) weights fp32 -> bf16
  k_cvt<<<(int)((DD + 255) / 256), 256, 0, stream>>>(qw, wq, (int)DD);
  k_cvt<<<(int)((DD + 255) / 256), 256, 0, stream>>>(kw, wk, (int)DD);
  k_cvt<<<(int)((DD + 255) / 256), 256, 0, stream>>>(vw, wv, (int)DD);
  k_cvt<<<(int)((DD + 255) / 256), 256, 0, stream>>>(ow, wo, (int)DD);
  k_cvt<<<(int)((EW + 255) / 256), 256, 0, stream>>>(egw, weg, (int)EW);
  k_cvt<<<(int)((EW + 255) / 256), 256, 0, stream>>>(euw, weu, (int)EW);
  k_cvt<<<(int)((EW + 255) / 256), 256, 0, stream>>>(edw, wed, (int)EW);

  // 2) rmsnorm1
  k_rmsnorm<<<N_, 256, 0, stream>>>(hidden, ln1, xn1);

  // 3) QKV projections (WMMA, wide tiles)
  dim3 gW(D_ / 128, N_ / 64);
  k_gemm_w<<<gW, 128, 0, stream>>>(xn1, D_, wq, D_, qf, nullptr, D_, nullptr, D_);
  k_gemm_w<<<gW, 128, 0, stream>>>(xn1, D_, wk, D_, kf, nullptr, D_, nullptr, D_);
  k_gemm_w<<<gW, 128, 0, stream>>>(xn1, D_, wv, D_, vf, nullptr, D_, nullptr, D_);

  // 4) RoPE + pack bf16 (Vt transposed)
  k_rope<<<dim3(N_, H_), 128, 0, stream>>>(qf, kf, vf, pos, qb, kb, vtb);

  // 5) flash attention
  k_flash<<<dim3(S_ / 64, B_ * H_), 128, 0, stream>>>(qb, kb, vtb, atb);

  // 6) O-projection + residual -> h
  k_gemm_w<<<gW, 128, 0, stream>>>(atb, D_, wo, D_, hf, nullptr, D_, hidden, D_);

  // 7) rmsnorm2 + router
  k_rmsnorm<<<N_, 256, 0, stream>>>(hf, ln2, xn2);
  k_router<<<N_, 256, 0, stream>>>(xn2, gatew, wgt);

  // 8) MoE gate/up fused (per-expert, skip unused tiles)
  k_moe_gu<<<dim3(DE_ / 64, N_ / 64, E_), 128, 0, stream>>>(xn2, weg, weu, wgt, inter);

  // 9) MoE down + weighted combine + residual -> out
  k_moe_down<<<dim3(D_ / 64, N_ / 64), 128, 0, stream>>>(inter, wed, wgt, hf, out);
}
